// SelfAttention_78512002171214
// MI455X (gfx1250) — compile-verified, matrix operand order checked by reading, not run on hardware
//
#include <hip/hip_runtime.h>
#include <hip/hip_bf16.h>
#include <stdint.h>

// ---------------------------------------------------------------------------
// Self-attention forward for MI455X (gfx1250), bf16 WMMA pipeline.
//   1) QKV projection GEMM (x fp32 -> Q/K bf16 row-major, V bf16 TRANSPOSED)
//   2) scores = Q K^T * scale (bf16 WMMA, fp32 accum) + per-row partial max
//   3) softmax: exp(s - rowmax) unnormalized (bf16) + fp32 row sums
//   4) O = P Vt^T (bf16 WMMA), epilogue divides by rowsum, fp32 out
// Every WMMA fragment (A and B side) is two aligned ds_load_b128 ops.
// Hot GEMM loops (2,4) stage tiles with GLOBAL_LOAD_ASYNC_TO_LDS_B128
// (ASYNCcnt-tracked DMA, double-buffered); kernel 1 keeps register staging
// because it converts fp32->bf16 in flight (v_cvt_pk_bf16_f32).
// ---------------------------------------------------------------------------

#define D_EMBED 1024
#define SEQ     4096
#define NBATCH  4
#define TOTROWS (NBATCH * SEQ)   // 16384
#define LSTR    40               // padded LDS row stride (ushorts): 80B = 20-bank skew

typedef __attribute__((ext_vector_type(16))) __bf16         v16bf;
typedef __attribute__((ext_vector_type(2)))  __bf16         v2bf;
typedef __attribute__((ext_vector_type(16))) unsigned short v16us;
typedef __attribute__((ext_vector_type(8)))  float          v8f;

struct U4x2 { uint4 a, b; };

// two fp32 -> packed bf16x2 dword (v_cvt_pk_bf16_f32)
__device__ __forceinline__ unsigned pack2bf(float lo, float hi) {
  v2bf t = {(__bf16)lo, (__bf16)hi};
  return __builtin_bit_cast(unsigned, t);
}
__device__ __forceinline__ unsigned short f2bf(float f) {
  return __builtin_bit_cast(unsigned short, (__bf16)f);
}
__device__ __forceinline__ float bf2f(unsigned short h) {
  unsigned u = ((unsigned)h) << 16;
  return __builtin_bit_cast(float, u);
}

__device__ __forceinline__ v8f wmma_bf16(v16us a, v16us b, v8f c) {
  // 8 args: (neg_a, A, neg_b, B, c_mod, C, reuse_a, reuse_b)
  return __builtin_amdgcn_wmma_f32_16x16x32_bf16(
      false, __builtin_bit_cast(v16bf, a),
      false, __builtin_bit_cast(v16bf, b),
      (short)0, c, false, false);
}

// ---- CDNA5 async global->LDS DMA ------------------------------------------
// 32-bit LDS offset of a __shared__ object (addrspace(3) cast).
__device__ __forceinline__ unsigned lds_addr32(const void* p) {
  return (unsigned)(uintptr_t)(__attribute__((address_space(3))) const char*)p;
}
__device__ __forceinline__ void async_b128(const void* lds_dst, const void* gsrc) {
  const unsigned l = lds_addr32(lds_dst);
  asm volatile("global_load_async_to_lds_b128 %0, %1, off"
               :: "v"(l), "v"(gsrc)
               : "memory");
}
__device__ __forceinline__ void wait_async0() {
#if __has_builtin(__builtin_amdgcn_s_wait_asynccnt)
  __builtin_amdgcn_s_wait_asynccnt(0);
#else
  asm volatile("s_wait_asynccnt 0x0" ::: "memory");
#endif
}

// Unified fragment loader: 16x32 bf16 tile, row-major in LDS (stride LSTR).
// Lane L<16: row L, K {0..7,16..23}; L>=16: row L-16, K {8..15,24..31}.
// Exactly two aligned 16B LDS loads, no repacking.
__device__ __forceinline__ v16us frag16x32(const unsigned short* tile, int row0) {
  const int lane = threadIdx.x & 31;
  const int r    = lane & 15;
  const int kh   = (lane >> 4) << 3;          // 0 or 8
  const unsigned short* p = tile + (size_t)(row0 + r) * LSTR + kh;
  U4x2 t;
  t.a = *(const uint4*)(p);                   // elems kh..kh+7
  t.b = *(const uint4*)(p + 16);              // elems 16+kh..16+kh+7
  return __builtin_bit_cast(v16us, t);
}

// ---- async staging: 128x32 bf16 tile from a row-major source --------------
// 256 threads x 2 passes x 16B = one 8KB tile, DMA'd straight into LDS.
__device__ __forceinline__ void async_rows(unsigned short (*dst)[LSTR],
                                           const unsigned short* src,
                                           size_t stride, int rbase, int k0) {
  const int tid = threadIdx.x;
  const int c4  = tid & 3;
#pragma unroll
  for (int p = 0; p < 2; ++p) {
    const int r = (tid >> 2) + p * 64;
    async_b128(&dst[r][c4 * 8],
               src + (size_t)(rbase + r) * stride + k0 + c4 * 8);
  }
}

// ---- register staging (kernel 1 only: fp32 -> bf16 conversion) ------------
__device__ __forceinline__ void ld_kmajor_f32(unsigned g[8], const float* src,
                                              int nbase, int k0) {
  const int tid = threadIdx.x;
  const int col = tid & 127;
  const int kh2 = (tid >> 7) * 16;
#pragma unroll
  for (int j = 0; j < 8; ++j) {
    const float lo = src[(size_t)(k0 + kh2 + 2 * j)     * D_EMBED + nbase + col];
    const float hi = src[(size_t)(k0 + kh2 + 2 * j + 1) * D_EMBED + nbase + col];
    g[j] = pack2bf(lo, hi);
  }
}
__device__ __forceinline__ void st_kmajor(unsigned short (*dst)[LSTR], const unsigned g[8]) {
  const int tid = threadIdx.x;
  const int col = tid & 127;
  const int kh2 = (tid >> 7) * 16;
  uint4 a = {g[0], g[1], g[2], g[3]};
  uint4 b = {g[4], g[5], g[6], g[7]};
  *(uint4*)&dst[col][kh2]     = a;
  *(uint4*)&dst[col][kh2 + 8] = b;
}

// ---------------------------------------------------------------------------
// Kernel 1: Q/K/V projection.  C[128x128] tile per block, K-chunks of 32.
// grid = (TOTROWS/128, D_EMBED/128, 3); block = 256 (8 waves, each 64x32).
// z==0/1 -> Q/K row-major [tok][d]; z==2 -> V transposed Vt[b][d][s]
// (in C layout each lane holds 8 consecutive s for one d -> one b128 store).
// ---------------------------------------------------------------------------
__global__ __launch_bounds__(256)
void qkv_project_kernel(const float* __restrict__ x,
                        const float* __restrict__ WQ,
                        const float* __restrict__ WK,
                        const float* __restrict__ WV,
                        unsigned short* __restrict__ Qb,
                        unsigned short* __restrict__ Kb,
                        unsigned short* __restrict__ Vt) {
  __shared__ unsigned short sA[2][128][LSTR];   // x tile (bf16)
  __shared__ unsigned short sB[2][128][LSTR];   // W tile, transposed [n][k]

  const int tid   = threadIdx.x;
  const int wave  = tid >> 5;
  const int lane  = tid & 31;
  const int mbase = blockIdx.x * 128;
  const int nbase = blockIdx.y * 128;
  const float* W = (blockIdx.z == 0) ? WQ : (blockIdx.z == 1) ? WK : WV;

  const int wm = (wave >> 2) * 64;   // 0 / 64
  const int wn = (wave & 3) * 32;    // 0..96

  const v8f vzero = {0.f,0.f,0.f,0.f,0.f,0.f,0.f,0.f};
  v8f acc[4][2];
#pragma unroll
  for (int i = 0; i < 4; ++i)
#pragma unroll
    for (int j = 0; j < 2; ++j) acc[i][j] = vzero;

  float4   ga[4];
  unsigned gb[8];
  const int ac = (tid & 7) * 4;      // A-staging column

  // prologue: chunk 0
  {
#pragma unroll
    for (int p = 0; p < 4; ++p) {
      const int r = (tid >> 3) + p * 32;
      ga[p] = *(const float4*)(x + (size_t)(mbase + r) * D_EMBED + ac);
    }
    ld_kmajor_f32(gb, W, nbase, 0);
#pragma unroll
    for (int p = 0; p < 4; ++p) {
      const int r = (tid >> 3) + p * 32;
      uint2 u = {pack2bf(ga[p].x, ga[p].y), pack2bf(ga[p].z, ga[p].w)};
      *(uint2*)&sA[0][r][ac] = u;
    }
    st_kmajor(sB[0], gb);
  }
  __syncthreads();

  for (int kt = 0; kt < D_EMBED / 32; ++kt) {
    const int  cur  = kt & 1;
    const bool more = (kt + 1 < D_EMBED / 32);
    if (more) {
      const int k0 = (kt + 1) * 32;
#pragma unroll
      for (int p = 0; p < 4; ++p) {
        const int r = (tid >> 3) + p * 32;
        ga[p] = *(const float4*)(x + (size_t)(mbase + r) * D_EMBED + k0 + ac);
      }
      ld_kmajor_f32(gb, W, nbase, k0);
    }
    const v16us fb0 = frag16x32(&sB[cur][0][0], wn);
    const v16us fb1 = frag16x32(&sB[cur][0][0], wn + 16);
#pragma unroll
    for (int mt = 0; mt < 4; ++mt) {
      const v16us fa = frag16x32(&sA[cur][0][0], wm + mt * 16);
      acc[mt][0] = wmma_bf16(fa, fb0, acc[mt][0]);
      acc[mt][1] = wmma_bf16(fa, fb1, acc[mt][1]);
    }
    if (more) {
#pragma unroll
      for (int p = 0; p < 4; ++p) {
        const int r = (tid >> 3) + p * 32;
        uint2 u = {pack2bf(ga[p].x, ga[p].y), pack2bf(ga[p].z, ga[p].w)};
        *(uint2*)&sA[cur ^ 1][r][ac] = u;
      }
      st_kmajor(sB[cur ^ 1], gb);
    }
    __syncthreads();
  }

  const int rofs = (lane >> 4) << 3;
  const int colL = lane & 15;
  if (blockIdx.z < 2) {
    // row-major [tok][d] store (C layout: VGPR r -> row r / r+8)
    unsigned short* dst = (blockIdx.z == 0) ? Qb : Kb;
#pragma unroll
    for (int mt = 0; mt < 4; ++mt)
#pragma unroll
      for (int nt = 0; nt < 2; ++nt)
#pragma unroll
        for (int r = 0; r < 8; ++r) {
          const int row = mbase + wm + mt * 16 + r + rofs;
          const int col = nbase + wn + nt * 16 + colL;
          dst[(size_t)row * D_EMBED + col] = f2bf(acc[mt][nt][r]);
        }
  } else {
    // V transposed: Vt[b][d][s]; lane's 8 values are consecutive s for one d
    const int b  = mbase >> 12;          // tiles never straddle batches
    const int s0 = mbase & (SEQ - 1);
    unsigned short* vt = Vt + (size_t)b * D_EMBED * SEQ;
#pragma unroll
    for (int mt = 0; mt < 4; ++mt)
#pragma unroll
      for (int nt = 0; nt < 2; ++nt) {
        const int d = nbase + wn + nt * 16 + colL;
        const int s = s0 + wm + mt * 16 + rofs;
        uint4 u = {pack2bf(acc[mt][nt][0], acc[mt][nt][1]),
                   pack2bf(acc[mt][nt][2], acc[mt][nt][3]),
                   pack2bf(acc[mt][nt][4], acc[mt][nt][5]),
                   pack2bf(acc[mt][nt][6], acc[mt][nt][7])};
        *(uint4*)(vt + (size_t)d * SEQ + s) = u;
      }
  }
}

// ---------------------------------------------------------------------------
// Kernel 2: scores tile S[128q x 128k] = Q K^T * scale for one batch.
// Each wave owns 16 query rows x 128 keys -> row stats stay in a 16-lane half.
// Tiles staged by async DMA (double-buffered).
// ---------------------------------------------------------------------------
__global__ __launch_bounds__(256)
void scores_kernel(const unsigned short* __restrict__ Q,
                   const unsigned short* __restrict__ K,
                   unsigned short* __restrict__ P,
                   float* __restrict__ partmax) {
  __shared__ unsigned short sQ[2][128][LSTR];
  __shared__ unsigned short sK[2][128][LSTR];

  const int tid   = threadIdx.x;
  const int wave  = tid >> 5;
  const int lane  = tid & 31;
  const int qbase = blockIdx.x * 128;
  const int kbase = blockIdx.y * 128;
  const float scale = 0.03125f;            // 1/sqrt(1024)

  const v8f vzero = {0.f,0.f,0.f,0.f,0.f,0.f,0.f,0.f};
  v8f acc[8];
#pragma unroll
  for (int i = 0; i < 8; ++i) acc[i] = vzero;

  async_rows(sQ[0], Q, D_EMBED, qbase, 0);
  async_rows(sK[0], K, D_EMBED, kbase, 0);
  wait_async0();
  __syncthreads();

  for (int kt = 0; kt < D_EMBED / 32; ++kt) {
    const int  cur  = kt & 1;
    const bool more = (kt + 1 < D_EMBED / 32);
    if (more) {
      async_rows(sQ[cur ^ 1], Q, D_EMBED, qbase, (kt + 1) * 32);
      async_rows(sK[cur ^ 1], K, D_EMBED, kbase, (kt + 1) * 32);
    }
    const v16us fq = frag16x32(&sQ[cur][0][0], wave * 16);
#pragma unroll
    for (int nt = 0; nt < 8; ++nt) {
      const v16us fk = frag16x32(&sK[cur][0][0], nt * 16);   // B = K^T
      acc[nt] = wmma_bf16(fq, fk, acc[nt]);
    }
    if (more) wait_async0();
    __syncthreads();
  }

  // epilogue: scale, bf16 store, per-row max across this 128-key tile
  float rmax[8];
#pragma unroll
  for (int r = 0; r < 8; ++r) rmax[r] = -3.0e38f;
  const int rofs = (lane >> 4) << 3;
  const int colL = lane & 15;
#pragma unroll
  for (int nt = 0; nt < 8; ++nt) {
#pragma unroll
    for (int r = 0; r < 8; ++r) {
      const float s = acc[nt][r] * scale;
      rmax[r] = fmaxf(rmax[r], s);
      const int row = qbase + wave * 16 + r + rofs;
      const int col = kbase + nt * 16 + colL;
      P[(size_t)row * SEQ + col] = f2bf(s);
    }
  }
  // reduce over the 16 lanes of this half (masks 1,2,4,8 stay in-half)
#pragma unroll
  for (int r = 0; r < 8; ++r) {
#pragma unroll
    for (int m = 1; m < 16; m <<= 1)
      rmax[r] = fmaxf(rmax[r], __shfl_xor(rmax[r], m, 32));
  }
  if ((lane & 15) == 0) {
#pragma unroll
    for (int r = 0; r < 8; ++r) {
      const int row = qbase + wave * 16 + r + rofs;
      partmax[(size_t)row * (SEQ / 128) + blockIdx.y] = rmax[r];
    }
  }
}

// ---------------------------------------------------------------------------
// Kernel 3: per-row softmax numerator. One 256-thread block per row.
// Writes unnormalized exp(s - max) in bf16 and the fp32 row sum.
// ---------------------------------------------------------------------------
__global__ __launch_bounds__(256)
void softmax_kernel(unsigned short* __restrict__ P,
                    const float* __restrict__ partmax,
                    float* __restrict__ rowsum) {
  const int row = blockIdx.x;
  const int tid = threadIdx.x;

  float m = -3.0e38f;
#pragma unroll
  for (int i = 0; i < SEQ / 128; ++i)
    m = fmaxf(m, partmax[(size_t)row * (SEQ / 128) + i]);

  unsigned short* prow = P + (size_t)row * SEQ;
  float s = 0.f;
#pragma unroll
  for (int j = 0; j < SEQ / (256 * 8); ++j) {
    const int base = (tid + j * 256) * 8;           // 8 bf16 per thread chunk
    uint4 u = *(const uint4*)(prow + base);
    unsigned w[4] = {u.x, u.y, u.z, u.w};
    unsigned o[4];
#pragma unroll
    for (int q = 0; q < 4; ++q) {
      const float lo = __expf(bf2f((unsigned short)(w[q] & 0xffffu)) - m);
      const float hi = __expf(bf2f((unsigned short)(w[q] >> 16)) - m);
      s += lo + hi;
      o[q] = pack2bf(lo, hi);
    }
    uint4 ov = {o[0], o[1], o[2], o[3]};
    *(uint4*)(prow + base) = ov;
  }
#pragma unroll
  for (int msk = 1; msk < 32; msk <<= 1) s += __shfl_xor(s, msk, 32);

  __shared__ float wsum[8];
  if ((tid & 31) == 0) wsum[tid >> 5] = s;
  __syncthreads();
  if (tid == 0) {
    float t = 0.f;
#pragma unroll
    for (int w8 = 0; w8 < 8; ++w8) t += wsum[w8];
    rowsum[row] = t;
  }
}

// ---------------------------------------------------------------------------
// Kernel 4: O[128x128] tile = P[4096-wide] @ V, epilogue * (1/rowsum), fp32.
// B tile comes straight from transposed Vt[d][s]; both tiles async-DMA'd.
// ---------------------------------------------------------------------------
__global__ __launch_bounds__(256)
void outgemm_kernel(const unsigned short* __restrict__ P,
                    const unsigned short* __restrict__ Vt,
                    const float* __restrict__ rowsum,
                    float* __restrict__ Out) {
  __shared__ unsigned short sP[2][128][LSTR];
  __shared__ unsigned short sV[2][128][LSTR];   // [n=d][k=key]

  const int tid   = threadIdx.x;
  const int wave  = tid >> 5;
  const int lane  = tid & 31;
  const int mbase = blockIdx.x * 128;
  const int nbase = blockIdx.y * 128;
  const int wm = (wave >> 2) * 64;
  const int wn = (wave & 3) * 32;

  const v8f vzero = {0.f,0.f,0.f,0.f,0.f,0.f,0.f,0.f};
  v8f acc[4][2];
#pragma unroll
  for (int i = 0; i < 4; ++i)
#pragma unroll
    for (int j = 0; j < 2; ++j) acc[i][j] = vzero;

  async_rows(sP[0], P,  SEQ, mbase, 0);
  async_rows(sV[0], Vt, SEQ, nbase, 0);
  wait_async0();
  __syncthreads();

  for (int kt = 0; kt < SEQ / 32; ++kt) {
    const int  cur  = kt & 1;
    const bool more = (kt + 1 < SEQ / 32);
    if (more) {
      async_rows(sP[cur ^ 1], P,  SEQ, mbase, (kt + 1) * 32);
      async_rows(sV[cur ^ 1], Vt, SEQ, nbase, (kt + 1) * 32);
    }
    const v16us fb0 = frag16x32(&sV[cur][0][0], wn);
    const v16us fb1 = frag16x32(&sV[cur][0][0], wn + 16);
#pragma unroll
    for (int mt = 0; mt < 4; ++mt) {
      const v16us fa = frag16x32(&sP[cur][0][0], wm + mt * 16);
      acc[mt][0] = wmma_bf16(fa, fb0, acc[mt][0]);
      acc[mt][1] = wmma_bf16(fa, fb1, acc[mt][1]);
    }
    if (more) wait_async0();
    __syncthreads();
  }

  const int rofs = (lane >> 4) << 3;
  const int colL = lane & 15;
#pragma unroll
  for (int mt = 0; mt < 4; ++mt)
#pragma unroll
    for (int r = 0; r < 8; ++r) {
      const int   row = mbase + wm + mt * 16 + r + rofs;
      const float ri  = 1.0f / rowsum[row];
#pragma unroll
      for (int nt = 0; nt < 2; ++nt) {
        const int col = nbase + wn + nt * 16 + colL;
        Out[(size_t)row * D_EMBED + col] = acc[mt][nt][r] * ri;
      }
    }
}

// ---------------------------------------------------------------------------
// Launcher. Workspace layout:
//   Qb/Kb bf16 [16384 x 1024], Vt bf16 [4][1024][4096] (3 x 32MB total),
//   P bf16 [4096 x 4096] (32MB, reused per batch),
//   partmax fp32 [4096 x 32], rowsum fp32 [4096].  Total ~128.6 MB.
// ---------------------------------------------------------------------------
extern "C" void kernel_launch(void* const* d_in, const int* in_sizes, int n_in,
                              void* d_out, int out_size, void* d_ws, size_t ws_size,
                              hipStream_t stream) {
  const float* x  = (const float*)d_in[0];
  const float* WQ = (const float*)d_in[1];
  const float* WK = (const float*)d_in[2];
  const float* WV = (const float*)d_in[3];
  float* out = (float*)d_out;

  const size_t qkvElems = (size_t)TOTROWS * D_EMBED;
  unsigned short* Qb = (unsigned short*)d_ws;
  unsigned short* Kb = Qb + qkvElems;
  unsigned short* Vt = Kb + qkvElems;
  unsigned short* P  = Vt + qkvElems;
  float* partmax = (float*)(P + (size_t)SEQ * SEQ);
  float* rowsum  = partmax + (size_t)SEQ * (SEQ / 128);

  dim3 blk(256);

  qkv_project_kernel<<<dim3(TOTROWS / 128, D_EMBED / 128, 3), blk, 0, stream>>>(
      x, WQ, WK, WV, Qb, Kb, Vt);

  for (int b = 0; b < NBATCH; ++b) {
    const unsigned short* Qbb = Qb + (size_t)b * SEQ * D_EMBED;
    const unsigned short* Kbb = Kb + (size_t)b * SEQ * D_EMBED;
    const unsigned short* Vtb = Vt + (size_t)b * SEQ * D_EMBED;
    float* outb = out + (size_t)b * SEQ * D_EMBED;

    scores_kernel<<<dim3(SEQ / 128, SEQ / 128), blk, 0, stream>>>(Qbb, Kbb, P, partmax);
    softmax_kernel<<<dim3(SEQ), blk, 0, stream>>>(P, partmax, rowsum);
    outgemm_kernel<<<dim3(SEQ / 128, D_EMBED / 128), blk, 0, stream>>>(P, Vtb, rowsum, outb);
  }
}
